// NemotronHMOE_25245817766294
// MI455X (gfx1250) — compile-verified, matrix operand order checked by reading, not run on hardware
//
#include <hip/hip_runtime.h>
#include <math.h>

// ---------------------------------------------------------------------------
// NemotronH MoE for MI455X (gfx1250).
// Pass 1: convert x + all weights f32->bf16 into workspace (HBM-stream, once).
// Pass 2: router (tiny), builds per-expert token lists.
// Pass 3/4: shared + routed expert GEMMs: bf16 WMMA hot loops, M-tile=32,
//           one B fragment shared by two v_wmma, software-pipelined fragment
//           loads (prefetch k+1 before WMMA of k), f32 atomic scatter.
// ---------------------------------------------------------------------------

typedef __bf16 bf16_t;
typedef __attribute__((ext_vector_type(16))) __bf16 v16bf;
typedef __attribute__((ext_vector_type(8)))  __bf16 v8bf;
typedef __attribute__((ext_vector_type(4)))  __bf16 v4bf;
typedef __attribute__((ext_vector_type(8)))  float  v8f;

#define T_TOKENS 4096
#define H_DIM    1024
#define E_NUM    16
#define I_DIM    512
#define IS_DIM   2048
#define MT       32                 // tokens per M tile (2 x 16-row WMMA tiles)
#define XSTR     (H_DIM + 8)        // padded LDS strides: conflict-free ds_load_b128
#define HSTR     (I_DIM + 8)
#define SMEM_BYTES (MT * XSTR * 2 + MT * HSTR * 2)   // 99,328 B -> 3 blocks/WGP

// ---- WMMA fragment helpers (ISA 7.12.2 layouts) ---------------------------

// A 16x32 bf16 from LDS row-major tile. lane<16: row=lane, K={0..7,16..23};
// lane>=16: row=lane-16, K={8..15,24..31}.  -> 2x ds_load_b128 per lane.
__device__ inline v16bf load_a_frag(const bf16_t* tile, int stride, int k0, int lane) {
    int row = lane & 15;
    int kb  = (lane < 16) ? 0 : 8;
    const bf16_t* p = tile + row * stride + k0 + kb;
    v16bf a;
#pragma unroll
    for (int j = 0; j < 8; ++j) a[j] = p[j];
#pragma unroll
    for (int j = 0; j < 8; ++j) a[8 + j] = p[16 + j];
    return a;
}

// B 32x16 bf16 from global: element (k,n) = src[(n0+n)*ld + k0+k]; each lane
// owns one output column, contiguous in k -> 2x global_load_b128 per lane.
__device__ inline v16bf load_b_frag(const bf16_t* __restrict__ src, int ld,
                                    int n0, int k0, int lane) {
    int col = lane & 15;
    int kb  = (lane < 16) ? 0 : 8;
    const bf16_t* p = src + (size_t)(n0 + col) * (size_t)ld + k0 + kb;
    v16bf b;
#pragma unroll
    for (int j = 0; j < 8; ++j) b[j] = p[j];
#pragma unroll
    for (int j = 0; j < 8; ++j) b[8 + j] = p[16 + j];
    return b;
}

__device__ inline v8f wmma_bf16(v16bf a, v16bf b, v8f c) {
    return __builtin_amdgcn_wmma_f32_16x16x32_bf16(false, a, false, b,
                                                   (short)0, c, false, false);
}

// Software-pipelined dual-row GEMM inner loop:
//   acc0/acc1 += A[rows 0..15 / 16..31] (LDS) x B (global), K = klen,
//   B column base n0, B k-offset koff.  Prefetches k+1 fragments before the
//   WMMAs of step k so global/ds latency overlaps matrix ops.
__device__ inline void gemm_dual(const bf16_t* __restrict__ T0,
                                 const bf16_t* __restrict__ T1, int tstr,
                                 const bf16_t* __restrict__ W, int wld,
                                 int n0, int koff, int klen,
                                 int lane, v8f& acc0, v8f& acc1) {
    v16bf b  = load_b_frag(W, wld, n0, koff, lane);
    v16bf a0 = load_a_frag(T0, tstr, 0, lane);
    v16bf a1 = load_a_frag(T1, tstr, 0, lane);
    for (int k0 = 32; k0 < klen; k0 += 32) {
        v16bf bn  = load_b_frag(W, wld, n0, koff + k0, lane);
        v16bf a0n = load_a_frag(T0, tstr, k0, lane);
        v16bf a1n = load_a_frag(T1, tstr, k0, lane);
        acc0 = wmma_bf16(a0, b, acc0);
        acc1 = wmma_bf16(a1, b, acc1);
        b = bn; a0 = a0n; a1 = a1n;
    }
    acc0 = wmma_bf16(a0, b, acc0);
    acc1 = wmma_bf16(a1, b, acc1);
}

// ---- f32 -> bf16 conversion pass (HBM streaming) --------------------------
__global__ __launch_bounds__(256)
void cvt_bf16_kernel(const float* __restrict__ src, bf16_t* __restrict__ dst, int n4) {
    int stride = gridDim.x * blockDim.x;
    for (int i = blockIdx.x * blockDim.x + threadIdx.x; i < n4; i += stride) {
        float4 v = ((const float4*)src)[i];
        v4bf o;
        o[0] = (bf16_t)v.x; o[1] = (bf16_t)v.y; o[2] = (bf16_t)v.z; o[3] = (bf16_t)v.w;
        ((v4bf*)dst)[i] = o;
    }
}

// ---- Router: one wave per token -------------------------------------------
__global__ __launch_bounds__(256)
void moe_router_kernel(const float* __restrict__ x, const float* __restrict__ rw,
                       const float* __restrict__ bias,
                       int* __restrict__ counts, int* __restrict__ list_tok,
                       float* __restrict__ list_w) {
    int wave = threadIdx.x >> 5;
    int lane = threadIdx.x & 31;
    int t = blockIdx.x * 8 + wave;

    int e    = lane & 15;
    int half = lane >> 4;
    const float* xt = x  + (size_t)t * H_DIM;
    const float* we = rw + (size_t)e * H_DIM;
    float s = 0.f;
    int h0 = half * (H_DIM / 2);
    for (int h = h0; h < h0 + H_DIM / 2; ++h) s += xt[h] * we[h];
    s += __shfl_xor(s, 16, 32);

    float lg[16];
#pragma unroll
    for (int i = 0; i < 16; ++i) lg[i] = __shfl(s, i, 32);

    if (lane == 0) {
        float scores[16], sfc[16];
#pragma unroll
        for (int i = 0; i < 16; ++i) {
            scores[i] = 1.f / (1.f + expf(-lg[i]));
            sfc[i]    = scores[i] + bias[i];
        }
        float gs[4];
#pragma unroll
        for (int g = 0; g < 4; ++g) {
            float m1 = -1e30f; int i1 = 0;
            for (int j = 0; j < 4; ++j)
                if (sfc[4 * g + j] > m1) { m1 = sfc[4 * g + j]; i1 = j; }
            float m2 = -1e30f;
            for (int j = 0; j < 4; ++j)
                if (j != i1 && sfc[4 * g + j] > m2) m2 = sfc[4 * g + j];
            gs[g] = m1 + m2;
        }
        int g1 = 0; float bm = gs[0];
        for (int g = 1; g < 4; ++g) if (gs[g] > bm) { bm = gs[g]; g1 = g; }
        int g2 = -1; float bm2 = -1e30f;
        for (int g = 0; g < 4; ++g)
            if (g != g1 && gs[g] > bm2) { bm2 = gs[g]; g2 = g; }
        float masked[16];
#pragma unroll
        for (int i = 0; i < 16; ++i) {
            int g = i >> 2;
            masked[i] = (g == g1 || g == g2) ? sfc[i] : 0.f;
        }
        int tidx[4]; float tw[4]; float sum = 0.f;
#pragma unroll
        for (int k = 0; k < 4; ++k) {
            float m = -1e30f; int im = 0;
            for (int i = 0; i < 16; ++i)
                if (masked[i] > m) { m = masked[i]; im = i; }
            masked[im] = -1e30f;
            tidx[k] = im;
            tw[k]   = scores[im];
            sum    += tw[k];
        }
        float scl = 2.5f / (sum + 1e-20f);
#pragma unroll
        for (int k = 0; k < 4; ++k) {
            int ex   = tidx[k];
            int slot = atomicAdd(&counts[ex], 1);
            list_tok[ex * T_TOKENS + slot] = t;
            list_w [ex * T_TOKENS + slot]  = tw[k] * scl;
        }
    }
}

// ---- Shared expert: atomic-accumulate relu2(X Wup^T) Wdown^T --------------
__global__ __launch_bounds__(256)
void moe_shared_kernel(const bf16_t* __restrict__ xb,
                       const bf16_t* __restrict__ wsub,
                       const bf16_t* __restrict__ wsdb,
                       float* __restrict__ out) {
    extern __shared__ char smem_raw[];
    bf16_t* Xt = (bf16_t*)smem_raw;                       // MT x XSTR
    bf16_t* Ht = (bf16_t*)(smem_raw + MT * XSTR * 2);     // MT x HSTR

    int tid = threadIdx.x;
    int t0  = blockIdx.x * MT;

    for (int idx = tid; idx < MT * (H_DIM / 8); idx += 256) {
        int r = idx >> 7;               // / (H_DIM/8)
        int c = (idx & 127) * 8;
        *(v8bf*)(Xt + r * XSTR + c) =
            *(const v8bf*)(xb + (size_t)(t0 + r) * H_DIM + c);
    }
    __syncthreads();

    int wave = tid >> 5, lane = tid & 31;
    int col = lane & 15, rbase = (lane < 16) ? 0 : 8;

    for (int ic = 0; ic < IS_DIM; ic += I_DIM) {          // 4 chunks of 512
        // GEMM1 chunk: relu2 -> bf16 H tile
        for (int nt = wave; nt < I_DIM / 16; nt += 8) {
            v8f acc0 = {}, acc1 = {};
            gemm_dual(Xt, Xt + 16 * XSTR, XSTR,
                      wsub, H_DIM, ic + nt * 16, 0, H_DIM, lane, acc0, acc1);
#pragma unroll
            for (int j = 0; j < 8; ++j) {
                float u0 = acc0[j]; u0 = (u0 > 0.f) ? u0 * u0 : 0.f;
                float u1 = acc1[j]; u1 = (u1 > 0.f) ? u1 * u1 : 0.f;
                Ht[(rbase + j) * HSTR + nt * 16 + col]      = (bf16_t)u0;
                Ht[(16 + rbase + j) * HSTR + nt * 16 + col] = (bf16_t)u1;
            }
        }
        __syncthreads();
        // GEMM2 partial over this i-chunk -> f32 atomic scatter (L2 resident)
        for (int nt = wave; nt < H_DIM / 16; nt += 8) {
            v8f acc0 = {}, acc1 = {};
            gemm_dual(Ht, Ht + 16 * HSTR, HSTR,
                      wsdb, IS_DIM, nt * 16, ic, I_DIM, lane, acc0, acc1);
#pragma unroll
            for (int j = 0; j < 8; ++j) {
                atomicAdd(&out[(size_t)(t0 + rbase + j) * H_DIM + nt * 16 + col], acc0[j]);
                atomicAdd(&out[(size_t)(t0 + 16 + rbase + j) * H_DIM + nt * 16 + col], acc1[j]);
            }
        }
        __syncthreads();
    }
}

// ---- Routed experts: gathered 32-token tiles ------------------------------
__global__ __launch_bounds__(256)
void moe_expert_kernel(const bf16_t* __restrict__ xb,
                       const bf16_t* __restrict__ wub,
                       const bf16_t* __restrict__ wdb,
                       const int* __restrict__ counts,
                       const int* __restrict__ list_tok,
                       const float* __restrict__ list_w,
                       float* __restrict__ out) {
    int e    = blockIdx.y;
    int tile = blockIdx.x;
    int n    = counts[e];
    if (tile * MT >= n) return;                           // uniform exit

    extern __shared__ char smem_raw[];
    bf16_t* Xt = (bf16_t*)smem_raw;
    bf16_t* Ht = (bf16_t*)(smem_raw + MT * XSTR * 2);
    __shared__ float cw[MT];
    __shared__ int   toks[MT];

    int tid = threadIdx.x;
    if (tid < MT) {
        int slot = tile * MT + tid;
        if (slot < n) {
            toks[tid] = list_tok[e * T_TOKENS + slot];
            cw[tid]   = list_w [e * T_TOKENS + slot];
        } else {                                          // pad: weight 0 => no effect
            toks[tid] = list_tok[e * T_TOKENS];
            cw[tid]   = 0.f;
        }
    }
    __syncthreads();

    for (int idx = tid; idx < MT * (H_DIM / 8); idx += 256) {
        int r = idx >> 7;
        int c = (idx & 127) * 8;
        *(v8bf*)(Xt + r * XSTR + c) =
            *(const v8bf*)(xb + (size_t)toks[r] * H_DIM + c);
    }
    __syncthreads();

    int wave = tid >> 5, lane = tid & 31;
    int col = lane & 15, rbase = (lane < 16) ? 0 : 8;
    const bf16_t* wu = wub + (size_t)e * I_DIM * H_DIM;
    const bf16_t* wd = wdb + (size_t)e * H_DIM * I_DIM;

    // GEMM1: relu2 * combine weight -> bf16 H tile
    for (int nt = wave; nt < I_DIM / 16; nt += 8) {
        v8f acc0 = {}, acc1 = {};
        gemm_dual(Xt, Xt + 16 * XSTR, XSTR,
                  wu, H_DIM, nt * 16, 0, H_DIM, lane, acc0, acc1);
#pragma unroll
        for (int j = 0; j < 8; ++j) {
            int r0 = rbase + j, r1 = 16 + rbase + j;
            float u0 = acc0[j]; u0 = (u0 > 0.f) ? u0 * u0 : 0.f;
            float u1 = acc1[j]; u1 = (u1 > 0.f) ? u1 * u1 : 0.f;
            Ht[r0 * HSTR + nt * 16 + col] = (bf16_t)(u0 * cw[r0]);
            Ht[r1 * HSTR + nt * 16 + col] = (bf16_t)(u1 * cw[r1]);
        }
    }
    __syncthreads();

    // GEMM2: f32 atomic scatter into out
    for (int nt = wave; nt < H_DIM / 16; nt += 8) {
        v8f acc0 = {}, acc1 = {};
        gemm_dual(Ht, Ht + 16 * HSTR, HSTR,
                  wd, I_DIM, nt * 16, 0, I_DIM, lane, acc0, acc1);
#pragma unroll
        for (int j = 0; j < 8; ++j) {
            atomicAdd(&out[(size_t)toks[rbase + j] * H_DIM + nt * 16 + col], acc0[j]);
            atomicAdd(&out[(size_t)toks[16 + rbase + j] * H_DIM + nt * 16 + col], acc1[j]);
        }
    }
}

// ---------------------------------------------------------------------------
extern "C" void kernel_launch(void* const* d_in, const int* in_sizes, int n_in,
                              void* d_out, int out_size, void* d_ws, size_t ws_size,
                              hipStream_t stream) {
    const float* x        = (const float*)d_in[0];
    const float* router_w = (const float*)d_in[1];
    const float* bias     = (const float*)d_in[2];
    const float* w_up     = (const float*)d_in[3];
    const float* w_down   = (const float*)d_in[4];
    const float* ws_up    = (const float*)d_in[5];
    const float* ws_down  = (const float*)d_in[6];
    float* out = (float*)d_out;

    // workspace layout
    char* ws = (char*)d_ws;
    size_t off = 0;
    int*   counts   = (int*)(ws + off);  off += 1024;
    int*   list_tok = (int*)(ws + off);  off += (size_t)E_NUM * T_TOKENS * 4;
    float* list_w   = (float*)(ws + off); off += (size_t)E_NUM * T_TOKENS * 4;
    bf16_t* xb   = (bf16_t*)(ws + off);  off += (size_t)T_TOKENS * H_DIM * 2;
    bf16_t* wub  = (bf16_t*)(ws + off);  off += (size_t)E_NUM * I_DIM * H_DIM * 2;
    bf16_t* wdb  = (bf16_t*)(ws + off);  off += (size_t)E_NUM * H_DIM * I_DIM * 2;
    bf16_t* wsub = (bf16_t*)(ws + off);  off += (size_t)IS_DIM * H_DIM * 2;
    bf16_t* wsdb = (bf16_t*)(ws + off);  off += (size_t)H_DIM * IS_DIM * 2;

    hipMemsetAsync(counts, 0, E_NUM * sizeof(int), stream);
    hipMemsetAsync(out, 0, (size_t)out_size * sizeof(float), stream);

    // one-shot f32->bf16 conversion (HBM streaming; bf16 images stay in L2)
    cvt_bf16_kernel<<<2048, 256, 0, stream>>>(x,       xb,   T_TOKENS * H_DIM / 4);
    cvt_bf16_kernel<<<2048, 256, 0, stream>>>(w_up,    wub,  E_NUM * I_DIM * H_DIM / 4);
    cvt_bf16_kernel<<<2048, 256, 0, stream>>>(w_down,  wdb,  E_NUM * H_DIM * I_DIM / 4);
    cvt_bf16_kernel<<<2048, 256, 0, stream>>>(ws_up,   wsub, IS_DIM * H_DIM / 4);
    cvt_bf16_kernel<<<2048, 256, 0, stream>>>(ws_down, wsdb, H_DIM * IS_DIM / 4);

    moe_router_kernel<<<T_TOKENS / 8, 256, 0, stream>>>(
        x, router_w, bias, counts, list_tok, list_w);

    moe_shared_kernel<<<T_TOKENS / MT, 256, SMEM_BYTES, stream>>>(
        xb, wsub, wsdb, out);

    moe_expert_kernel<<<dim3(T_TOKENS / MT, E_NUM), 256, SMEM_BYTES, stream>>>(
        xb, wub, wdb, counts, list_tok, list_w, out);
}
